// MistralDecoderLayer_73933567034007
// MI455X (gfx1250) — matrix operand order repaired and checked by reading, not compiled, not run on hardware
//
#include <hip/hip_runtime.h>

#define SEQ 2048
#define HID 4096
#define NHEAD 32
#define HDIM 128
#define FF 14336

typedef __attribute__((ext_vector_type(16))) __bf16 v16bf;
typedef __attribute__((ext_vector_type(8)))  float  v8f;
typedef unsigned short u16;

static __device__ __forceinline__ v8f wmma_bf16(v16bf a, v16bf b, v8f c) {
  // D(f32 16x16) = A(bf16 16x32) x B(bf16 32x16) + C
  return __builtin_amdgcn_wmma_f32_16x16x32_bf16(false, a, false, b, (short)0, c,
                                                 false, false);
}

// CDNA5 async global->LDS copy (ASYNCcnt-tracked), 16 bytes per lane.
// LDS operand: wave-relative LDS offset = low 32 bits of the generic pointer
// (ISA 10.2: flat LDS addresses truncate to LDS offset).
static __device__ __forceinline__ void async_copy_b128(void* lds_dst,
                                                       const void* gsrc) {
  unsigned lds = (unsigned)(size_t)lds_dst;
  unsigned long long ga = (unsigned long long)(size_t)gsrc;
  asm volatile("global_load_async_to_lds_b128 %0, %1, off"
               :: "v"(lds), "v"(ga) : "memory");
}
static __device__ __forceinline__ void wait_async0() {
  asm volatile("s_wait_asynccnt 0x0" ::: "memory");
}

// ---------------------------------------------------------------------------
// RMSNorm: fp32 in -> bf16 out (one block per row)
// ---------------------------------------------------------------------------
__global__ __launch_bounds__(256) void rmsnorm_bf16_kernel(
    const float* __restrict__ x, const float* __restrict__ w,
    u16* __restrict__ outb, int n) {
  const int row = blockIdx.x;
  const float* xr = x + (size_t)row * n;
  __bf16* ob = reinterpret_cast<__bf16*>(outb) + (size_t)row * n;
  float ss = 0.f;
  for (int c = threadIdx.x; c < n; c += 256) { float v = xr[c]; ss += v * v; }
  #pragma unroll
  for (int m = 16; m >= 1; m >>= 1) ss += __shfl_xor(ss, m, 32);
  __shared__ float red[8];
  if ((threadIdx.x & 31) == 0) red[threadIdx.x >> 5] = ss;
  __syncthreads();
  float tot = 0.f;
  #pragma unroll
  for (int i = 0; i < 8; ++i) tot += red[i];
  const float rs = rsqrtf(tot / (float)n + 1e-6f);
  for (int c = threadIdx.x; c < n; c += 256)
    ob[c] = (__bf16)(xr[c] * rs * w[c]);
}

// ---------------------------------------------------------------------------
// Generic WMMA GEMM: C[M,N] = A_bf16[M,K] * W_fp32[N,K]^T (+bias)(+resid)
// 64x64 tile / 256 threads (8 waves, each 16x32 via two 16x16 WMMA frags).
// A tile: async global->LDS (ASYNCcnt). W tile: NT fp32 loads, convert-on-
// stage to bf16 (weights stream once; keep L2 for reused activations).
// ---------------------------------------------------------------------------
template<bool OUT_BF16, bool HAS_BIAS, bool HAS_RESID>
__global__ __launch_bounds__(256) void gemm_bf16_wmma(
    const u16* __restrict__ Au, const float* __restrict__ W,
    const float* __restrict__ bias, const float* __restrict__ resid,
    void* __restrict__ Cout, int M, int N, int K) {
  const __bf16* A = reinterpret_cast<const __bf16*>(Au);
  __shared__ alignas(16) __bf16 As[64][32];
  __shared__ alignas(16) __bf16 Bs[64][32];
  const int tid  = threadIdx.x;
  const int lane = tid & 31;
  const int wv   = tid >> 5;
  const int m0 = blockIdx.y * 64, n0 = blockIdx.x * 64;
  const int moff = (wv & 3) * 16, noff = (wv >> 2) * 32;
  const int lg = (lane >> 4) & 1, lm = lane & 15;
  v8f acc0 = {}, acc1 = {};

  const int lr = (tid * 8) >> 5, lc = (tid * 8) & 31;  // tile load coords
  for (int kt = 0; kt < K; kt += 32) {
    // A tile: 64x32 bf16, async copy straight into LDS
    async_copy_b128(&As[lr][lc], &A[(size_t)(m0 + lr) * K + kt + lc]);
    // W tile: 64x32 fp32 -> bf16 convert-on-stage, non-temporal stream
    {
      const float* wsrc = &W[(size_t)(n0 + lr) * K + kt + lc];
      #pragma unroll
      for (int j = 0; j < 8; ++j)
        Bs[lr][lc + j] = (__bf16)__builtin_nontemporal_load(wsrc + j);
      if (kt + 32 < K) __builtin_prefetch(wsrc + 32, 0, 3);  // WGP-scope prefetch
    }
    wait_async0();
    __syncthreads();
    v16bf af, bf0, bf1;
    #pragma unroll
    for (int i = 0; i < 16; ++i) {       // A frag: 16x32, contiguous k = i + 16*lg
      const int kk = i + lg * 16;
      af[i] = As[moff + lm][kk];
    }
    #pragma unroll
    for (int i = 0; i < 16; ++i) {       // B frags: 32x16, same contiguous k = i + 16*lg
      const int kk = i + lg * 16;
      bf0[i] = Bs[noff + lm][kk];
      bf1[i] = Bs[noff + 16 + lm][kk];
    }
    acc0 = wmma_bf16(af, bf0, acc0);
    acc1 = wmma_bf16(af, bf1, acc1);
    __syncthreads();
  }
  // Epilogue: C layout m = i + 8*lg, n = lane&15
  #pragma unroll
  for (int i = 0; i < 8; ++i) {
    const int m = m0 + moff + i + lg * 8;
    #pragma unroll
    for (int f = 0; f < 2; ++f) {
      const int n = n0 + noff + f * 16 + lm;
      float v = (f == 0) ? acc0[i] : acc1[i];
      if constexpr (HAS_BIAS)  v += bias[n];
      if constexpr (HAS_RESID) v += resid[(size_t)m * N + n];
      if constexpr (OUT_BF16)
        reinterpret_cast<__bf16*>(Cout)[(size_t)m * N + n] = (__bf16)v;
      else
        reinterpret_cast<float*>(Cout)[(size_t)m * N + n] = v;
    }
  }
}

// ---------------------------------------------------------------------------
// Fused gate/up projection: out = silu(A*Wg^T) * (A*Wu^T), bf16 out.
// ---------------------------------------------------------------------------
__global__ __launch_bounds__(256) void gemm_gateup_wmma(
    const u16* __restrict__ Au, const float* __restrict__ Wg,
    const float* __restrict__ Wu, u16* __restrict__ Cout, int M, int N, int K) {
  const __bf16* A = reinterpret_cast<const __bf16*>(Au);
  __shared__ alignas(16) __bf16 As[64][32];
  __shared__ alignas(16) __bf16 Gs[64][32];
  __shared__ alignas(16) __bf16 Us[64][32];
  const int tid = threadIdx.x, lane = tid & 31, wv = tid >> 5;
  const int m0 = blockIdx.y * 64, n0 = blockIdx.x * 64;
  const int moff = (wv & 3) * 16, noff = (wv >> 2) * 32;
  const int lg = (lane >> 4) & 1, lm = lane & 15;
  v8f ag0 = {}, ag1 = {}, au0 = {}, au1 = {};
  const int lr = (tid * 8) >> 5, lc = (tid * 8) & 31;
  for (int kt = 0; kt < K; kt += 32) {
    async_copy_b128(&As[lr][lc], &A[(size_t)(m0 + lr) * K + kt + lc]);
    {
      const float* g = &Wg[(size_t)(n0 + lr) * K + kt + lc];
      const float* u = &Wu[(size_t)(n0 + lr) * K + kt + lc];
      #pragma unroll
      for (int j = 0; j < 8; ++j) {
        Gs[lr][lc + j] = (__bf16)__builtin_nontemporal_load(g + j);
        Us[lr][lc + j] = (__bf16)__builtin_nontemporal_load(u + j);
      }
      if (kt + 32 < K) { __builtin_prefetch(g + 32, 0, 3); __builtin_prefetch(u + 32, 0, 3); }
    }
    wait_async0();
    __syncthreads();
    v16bf af, g0, g1, u0, u1;
    #pragma unroll
    for (int i = 0; i < 16; ++i) {
      const int kk = i + lg * 16;
      af[i] = As[moff + lm][kk];
    }
    #pragma unroll
    for (int i = 0; i < 16; ++i) {
      const int kk = i + lg * 16;
      g0[i] = Gs[noff + lm][kk];      g1[i] = Gs[noff + 16 + lm][kk];
      u0[i] = Us[noff + lm][kk];      u1[i] = Us[noff + 16 + lm][kk];
    }
    ag0 = wmma_bf16(af, g0, ag0);
    ag1 = wmma_bf16(af, g1, ag1);
    au0 = wmma_bf16(af, u0, au0);
    au1 = wmma_bf16(af, u1, au1);
    __syncthreads();
  }
  __bf16* C = reinterpret_cast<__bf16*>(Cout);
  #pragma unroll
  for (int i = 0; i < 8; ++i) {
    const int m = m0 + moff + i + lg * 8;
    #pragma unroll
    for (int f = 0; f < 2; ++f) {
      const int n = n0 + noff + f * 16 + lm;
      const float g = (f == 0) ? ag0[i] : ag1[i];
      const float u = (f == 0) ? au0[i] : au1[i];
      const float s = g / (1.f + __expf(-g));      // silu
      C[(size_t)m * N + n] = (__bf16)(s * u);
    }
  }
}

// ---------------------------------------------------------------------------
// Flash attention (causal), bf16 Q/K/V, bf16 out. Block = (64 q-rows, 1 head),
// 128 threads = 4 waves, each wave owns 16 q-rows x full d=128.
// K/V tiles stream through LDS via the async pipe; WMMA for QK^T and PV.
// ---------------------------------------------------------------------------
__global__ __launch_bounds__(128) void flash_attn_wmma(
    const u16* __restrict__ Qu, const u16* __restrict__ Ku,
    const u16* __restrict__ Vu, u16* __restrict__ Ou) {
  const __bf16* Q = reinterpret_cast<const __bf16*>(Qu);
  const __bf16* K = reinterpret_cast<const __bf16*>(Ku);
  const __bf16* V = reinterpret_cast<const __bf16*>(Vu);
  __bf16* O = reinterpret_cast<__bf16*>(Ou);
  __shared__ alignas(16) __bf16 Ks[32][HDIM];
  __shared__ alignas(16) __bf16 Vs[32][HDIM];
  __shared__ alignas(16) __bf16 Ps[4][16][32];
  const int tid = threadIdx.x, lane = tid & 31, wv = tid >> 5;
  const int h = blockIdx.y;
  const int q0 = blockIdx.x * 64;
  const int moff = wv * 16;
  const int lg = (lane >> 4) & 1, lm = lane & 15;
  const float scale = 0.088388347648318447f;  // 1/sqrt(128)

  // Q fragments for this wave's 16 rows: 4 chunks of K=32 (d = 0..127)
  v16bf qf[4];
  {
    const __bf16* qrow = Q + (size_t)(q0 + moff + lm) * (NHEAD * HDIM) + h * HDIM;
    #pragma unroll
    for (int c = 0; c < 4; ++c)
      #pragma unroll
      for (int i = 0; i < 16; ++i) {
        const int kk = c * 32 + i + lg * 16;
        qf[c][i] = qrow[kk];
      }
  }
  v8f of[8] = {};            // O accumulators: 8 d-chunks of 16
  float mrow[8], lrow[8];
  #pragma unroll
  for (int i = 0; i < 8; ++i) { mrow[i] = -3.0e38f; lrow[i] = 0.f; }

  const int nkb = (q0 + 64) / 32;  // causal: only key blocks <= q range
  for (int kb = 0; kb < nkb; ++kb) {
    {  // async K/V tile load: 32 keys x 128 d, 32 bf16 per thread (4x b128)
      const int r = tid >> 2, d0 = (tid & 3) * 32;
      const size_t base = (size_t)(kb * 32 + r) * (NHEAD * HDIM) + h * HDIM + d0;
      #pragma unroll
      for (int j = 0; j < 4; ++j) {
        async_copy_b128(&Ks[r][d0 + j * 8], K + base + j * 8);
        async_copy_b128(&Vs[r][d0 + j * 8], V + base + j * 8);
      }
    }
    wait_async0();
    __syncthreads();

    // S = Q * K^T  (16x32 per wave as two 16x16 frags)
    v8f sf0 = {}, sf1 = {};
    #pragma unroll
    for (int c = 0; c < 4; ++c) {
      v16bf b0, b1;
      #pragma unroll
      for (int i = 0; i < 16; ++i) {
        const int kk = c * 32 + i + lg * 16;
        b0[i] = Ks[lm][kk];        // keys 0..15 of block
        b1[i] = Ks[16 + lm][kk];   // keys 16..31 of block
      }
      sf0 = wmma_bf16(qf[c], b0, sf0);
      sf1 = wmma_bf16(qf[c], b1, sf1);
    }

    // scale + causal mask + row max (rows i correspond to m = moff+i+8*lg)
    const bool needmask = (kb + 2 >= nkb);
    float rm[8];
    #pragma unroll
    for (int i = 0; i < 8; ++i) {
      const int qabs = q0 + moff + i + lg * 8;
      float a0 = sf0[i] * scale, a1 = sf1[i] * scale;
      if (needmask) {
        if (kb * 32 + lm > qabs)      a0 = -3.0e38f;
        if (kb * 32 + 16 + lm > qabs) a1 = -3.0e38f;
      }
      sf0[i] = a0; sf1[i] = a1;
      rm[i] = fmaxf(a0, a1);
    }
    #pragma unroll
    for (int i = 0; i < 8; ++i) {   // reduce over the 16-lane column group
      rm[i] = fmaxf(rm[i], __shfl_xor(rm[i], 1, 32));
      rm[i] = fmaxf(rm[i], __shfl_xor(rm[i], 2, 32));
      rm[i] = fmaxf(rm[i], __shfl_xor(rm[i], 4, 32));
      rm[i] = fmaxf(rm[i], __shfl_xor(rm[i], 8, 32));
    }
    float corr[8], rs[8];
    #pragma unroll
    for (int i = 0; i < 8; ++i) {
      const float mn = fmaxf(mrow[i], rm[i]);
      corr[i] = __expf(mrow[i] - mn);
      mrow[i] = mn;
      const float p0 = __expf(sf0[i] - mn);
      const float p1 = __expf(sf1[i] - mn);
      sf0[i] = p0; sf1[i] = p1;
      rs[i] = p0 + p1;
    }
    #pragma unroll
    for (int i = 0; i < 8; ++i) {
      rs[i] += __shfl_xor(rs[i], 1, 32);
      rs[i] += __shfl_xor(rs[i], 2, 32);
      rs[i] += __shfl_xor(rs[i], 4, 32);
      rs[i] += __shfl_xor(rs[i], 8, 32);
      lrow[i] = lrow[i] * corr[i] + rs[i];
    }
    #pragma unroll
    for (int n = 0; n < 8; ++n)
      #pragma unroll
      for (int i = 0; i < 8; ++i) of[n][i] *= corr[i];

    // transpose P (C layout) -> A layout through per-wave LDS scratch
    #pragma unroll
    for (int i = 0; i < 8; ++i) {
      const int r = i + lg * 8;
      Ps[wv][r][lm]      = (__bf16)sf0[i];
      Ps[wv][r][16 + lm] = (__bf16)sf1[i];
    }
    __syncthreads();
    v16bf pf;
    #pragma unroll
    for (int i = 0; i < 16; ++i) {
      const int kk = i + lg * 16;
      pf[i] = Ps[wv][lm][kk];
    }
    // O += P(16x32) * V(32x16) per d-chunk
    #pragma unroll
    for (int n = 0; n < 8; ++n) {
      v16bf vb;
      #pragma unroll
      for (int i = 0; i < 16; ++i)
        vb[i] = Vs[i + lg * 16][n * 16 + lm];
      of[n] = wmma_bf16(pf, vb, of[n]);
    }
    __syncthreads();
  }
  #pragma unroll
  for (int i = 0; i < 8; ++i) {
    const int row = q0 + moff + i + lg * 8;
    const float inv = 1.f / lrow[i];
    #pragma unroll
    for (int n = 0; n < 8; ++n)
      O[(size_t)row * (NHEAD * HDIM) + h * HDIM + n * 16 + lm] =
          (__bf16)(of[n][i] * inv);
  }
}

// ---------------------------------------------------------------------------
// Host orchestration
// ---------------------------------------------------------------------------
extern "C" void kernel_launch(void* const* d_in, const int* in_sizes, int n_in,
                              void* d_out, int out_size, void* d_ws, size_t ws_size,
                              hipStream_t stream) {
  const float* hidden = (const float*)d_in[0];
  // d_in[1] = position_ids (unused: reference applies no RoPE)
  const float* wq = (const float*)d_in[2];  const float* bq = (const float*)d_in[3];
  const float* wk = (const float*)d_in[4];  const float* bk = (const float*)d_in[5];
  const float* wv = (const float*)d_in[6];  const float* bv = (const float*)d_in[7];
  const float* wo = (const float*)d_in[8];  const float* bo = (const float*)d_in[9];
  const float* wg = (const float*)d_in[10];
  const float* wu = (const float*)d_in[11];
  const float* wd = (const float*)d_in[12];
  const float* ln1 = (const float*)d_in[13];
  const float* ln2 = (const float*)d_in[14];
  float* out = (float*)d_out;

  char* ws = (char*)d_ws;
  const size_t SH = (size_t)SEQ * HID;
  size_t off = 0;
  u16* xb  = (u16*)(ws + off); off += SH * 2;               // normed x / y (bf16)
  u16* qb  = (u16*)(ws + off); off += SH * 2;               // Q (bf16)
  u16* kb  = (u16*)(ws + off); off += SH * 2;               // K (bf16)
  u16* vb  = (u16*)(ws + off); off += SH * 2;               // V (bf16)
  u16* ab  = (u16*)(ws + off); off += SH * 2;               // attn out (bf16)
  float* h1 = (float*)(ws + off); off += SH * 4;            // residual 1 (fp32)
  u16* act = (u16*)(ws + off); off += (size_t)SEQ * FF * 2; // silu(g)*u (bf16)

  const dim3 blk256(256), blk128(128);
  const dim3 gH(HID / 64, SEQ / 64);   // (64, 32)
  const dim3 gF(FF / 64, SEQ / 64);    // (224, 32)

  // 1) RMSNorm -> bf16
  rmsnorm_bf16_kernel<<<dim3(SEQ), blk256, 0, stream>>>(hidden, ln1, xb, HID);
  // 2) Q, K, V projections (bias, bf16 out)
  gemm_bf16_wmma<true, true, false><<<gH, blk256, 0, stream>>>(xb, wq, bq, nullptr, qb, SEQ, HID, HID);
  gemm_bf16_wmma<true, true, false><<<gH, blk256, 0, stream>>>(xb, wk, bk, nullptr, kb, SEQ, HID, HID);
  gemm_bf16_wmma<true, true, false><<<gH, blk256, 0, stream>>>(xb, wv, bv, nullptr, vb, SEQ, HID, HID);
  // 3) causal flash attention
  flash_attn_wmma<<<dim3(SEQ / 64, NHEAD), blk128, 0, stream>>>(qb, kb, vb, ab);
  // 4) O projection + bias + residual -> h1 (fp32)
  gemm_bf16_wmma<false, true, true><<<gH, blk256, 0, stream>>>(ab, wo, bo, hidden, h1, SEQ, HID, HID);
  // 5) RMSNorm(h1) -> bf16
  rmsnorm_bf16_kernel<<<dim3(SEQ), blk256, 0, stream>>>(h1, ln2, xb, HID);
  // 6) fused gate/up -> silu(g)*u (bf16)
  gemm_gateup_wmma<<<gF, blk256, 0, stream>>>(xb, wg, wu, act, SEQ, FF, HID);
  // 7) down projection + residual -> out (fp32)
  gemm_bf16_wmma<false, false, true><<<gH, blk256, 0, stream>>>(act, wd, nullptr, h1, out, SEQ, HID, FF);
}
